// Temper_27599459844279
// MI455X (gfx1250) — matrix operand order repaired and checked by reading, not compiled, not run on hardware
//
#include <hip/hip_runtime.h>
#include <hip/hip_bf16.h>
#include <stdint.h>

#define BATCH   16384
#define HID     1024
#define EMBD    512
#define NOPS    8
#define K1      (HID + EMBD)   // 1536
#define BM      128
#define BN      128
#define KC      64
#define MAX_TILES 136          // 16384/128 + 8 partial tiles

typedef __attribute__((ext_vector_type(16))) __bf16 v16bf;
typedef __attribute__((ext_vector_type(8)))  float  v8f;

union FragBF { v16bf bf; uint4 q[2]; };

__device__ __forceinline__ unsigned short f2bf(float f) {
    unsigned int u = __float_as_uint(f);
    u += 0x7FFFu + ((u >> 16) & 1u);        // round-to-nearest-even
    return (unsigned short)(u >> 16);
}

// ---------------- routing ----------------
__global__ void k_zero_counts(int* counts) {
    if (threadIdx.x < NOPS) counts[threadIdx.x] = 0;
}

__global__ void k_hist(const int* __restrict__ ops, int* __restrict__ counts) {
    int i = blockIdx.x * blockDim.x + threadIdx.x;
    if (i < BATCH) atomicAdd(&counts[ops[i]], 1);
}

__global__ void k_build(const int* __restrict__ counts, int* __restrict__ cursor,
                        int4* __restrict__ tdesc, int* __restrict__ ntiles) {
    if (threadIdx.x != 0) return;
    int off = 0, nt = 0;
    for (int o = 0; o < NOPS; ++o) {
        int c = counts[o];
        cursor[o] = off;
        for (int t = 0; t * BM < c; ++t) {
            int rows = c - t * BM; if (rows > BM) rows = BM;
            tdesc[nt++] = make_int4(o, off + t * BM, rows, 0);
        }
        off += c;
    }
    *ntiles = nt;
}

__global__ void k_scatter(const int* __restrict__ ops, int* __restrict__ cursor,
                          int* __restrict__ stok) {
    int i = blockIdx.x * blockDim.x + threadIdx.x;
    if (i < BATCH) { int pos = atomicAdd(&cursor[ops[i]], 1); stok[pos] = i; }
}

// ---------------- weight / embedding conversion ----------------
__global__ void k_cvt_emb(const float* __restrict__ e, unsigned short* __restrict__ d) {
    int i = blockIdx.x * blockDim.x + threadIdx.x;
    if (i < NOPS * EMBD) d[i] = f2bf(e[i]);
}

// src: [op][K][N] fp32 row-major  ->  dst: [op][N][K] bf16 (column-major view)
__global__ void k_transpose_cvt(const float* __restrict__ src,
                                unsigned short* __restrict__ dst, int K, int N) {
    __shared__ float tile[32][33];
    int op = blockIdx.z;
    const float* s = src + (size_t)op * K * N;
    unsigned short* d = dst + (size_t)op * N * K;
    int n0 = blockIdx.x * 32, k0 = blockIdx.y * 32;
    int tx = threadIdx.x, ty = threadIdx.y;
#pragma unroll
    for (int i = 0; i < 32; i += 8)
        tile[ty + i][tx] = s[(size_t)(k0 + ty + i) * N + (n0 + tx)];
    __syncthreads();
#pragma unroll
    for (int i = 0; i < 32; i += 8)
        d[(size_t)(n0 + ty + i) * K + (k0 + tx)] = f2bf(tile[tx][ty + i]);
}

// ---------------- grouped expert GEMM ----------------
// 256 threads = 8 waves; block tile 128x128; 64-wide bf16 K panels,
// double-buffered in LDS with register prefetch of the next panel.
template<int KDIM, bool LAYER1>
__global__ __launch_bounds__(256)
void k_expert_gemm(const float* __restrict__ x,
                   const unsigned short* __restrict__ embb,
                   const unsigned short* __restrict__ hsrc,
                   const unsigned short* __restrict__ wt,      // [op][N=1024][KDIM] bf16
                   const float* __restrict__ bias,             // [op][1024]
                   const int* __restrict__ stok,
                   const int4* __restrict__ tdesc,
                   const int* __restrict__ ntiles,
                   unsigned short* __restrict__ hout,
                   float* __restrict__ yout) {
    if ((int)blockIdx.x >= *ntiles) return;
    int4 td = tdesc[blockIdx.x];
    const int expert = td.x, rowStart = td.y, rows = td.z;
    const int n0blk = blockIdx.y * BN;

    // padded strides (72 shorts = 144B) -> conflict-free b128 LDS reads
    __shared__ unsigned short As[2][BM][72];
    __shared__ unsigned short Bs[2][BN][72];

    const int tid  = threadIdx.x;
    const int lane = tid & 31, wid = tid >> 5;
    const int cl   = lane & 15, hl = lane >> 4;
    const int m0   = wid * 16;

    // staging coordinates: each thread handles 32 bf16 of one row/column
    const int sr = tid >> 1;                    // 0..127
    const int sh = tid & 1;                     // K-half within 64-wide panel
    const int rclamp  = sr < rows ? sr : rows - 1;
    const int stokIdx = stok[rowStart + rclamp];
    const unsigned short* wrow =
        wt + (size_t)expert * HID * KDIM + (size_t)(n0blk + sr) * KDIM;

    // register prefetch state for the next K panel
    float4 rf[8];   // LAYER1 fp32 x panel
    uint4  ra[4];   // bf16 A panel (emb / h)
    uint4  rb[4];   // bf16 B panel (weights)

    auto prefetch = [&](int kbv) {
        if (LAYER1) {
            if (kbv < HID) {
                const float4* p = (const float4*)(x + (size_t)stokIdx * HID + kbv + 32 * sh);
#pragma unroll
                for (int i = 0; i < 8; ++i) rf[i] = p[i];
            } else {
                const uint4* p = (const uint4*)(embb + (size_t)expert * EMBD +
                                                (kbv - HID) + 32 * sh);
#pragma unroll
                for (int i = 0; i < 4; ++i) ra[i] = p[i];
            }
        } else {
            const uint4* p = (const uint4*)(hsrc + (size_t)stokIdx * HID + kbv + 32 * sh);
#pragma unroll
            for (int i = 0; i < 4; ++i) ra[i] = p[i];
        }
        const uint4* q = (const uint4*)(wrow + kbv + 32 * sh);
#pragma unroll
        for (int i = 0; i < 4; ++i) rb[i] = q[i];
    };

    auto commit = [&](int bufIdx, int kbv) {
        uint4 tq[4];
        if (LAYER1 && kbv < HID) {
            unsigned int* tw = (unsigned int*)tq;
#pragma unroll
            for (int i = 0; i < 8; ++i) {
                tw[2 * i]     = ((unsigned int)f2bf(rf[i].y) << 16) | f2bf(rf[i].x);
                tw[2 * i + 1] = ((unsigned int)f2bf(rf[i].w) << 16) | f2bf(rf[i].z);
            }
        } else {
#pragma unroll
            for (int i = 0; i < 4; ++i) tq[i] = ra[i];
        }
#pragma unroll
        for (int i = 0; i < 4; ++i)
            *(uint4*)&As[bufIdx][sr][32 * sh + 8 * i] = tq[i];
#pragma unroll
        for (int i = 0; i < 4; ++i)
            *(uint4*)&Bs[bufIdx][sr][32 * sh + 8 * i] = rb[i];
    };

    const v8f vzero = {0.f, 0.f, 0.f, 0.f, 0.f, 0.f, 0.f, 0.f};
    v8f acc[8];
#pragma unroll
    for (int n = 0; n < 8; ++n) acc[n] = vzero;

    prefetch(0);
    commit(0, 0);
    __syncthreads();

    int buf = 0;
    for (int kb = 0; kb < KDIM; kb += KC) {
        const int nkb = kb + KC;
        const bool have_next = nkb < KDIM;
        if (have_next) prefetch(nkb);          // global loads overlap compute below

#pragma unroll
        for (int k0l = 0; k0l < KC; k0l += 32) {
            // A frag: lanes 0-15 rows m0..m0+15 K{0-7,16-23}; lanes 16-31 K{8-15,24-31}
            FragBF fa;
            fa.q[0] = *(const uint4*)&As[buf][m0 + cl][k0l + 8 * hl];
            fa.q[1] = *(const uint4*)&As[buf][m0 + cl][k0l + 16 + 8 * hl];
            // preload all 8 B frags so WMMAs can issue back-to-back
            FragBF fb[8];
#pragma unroll
            for (int n = 0; n < 8; ++n) {
                const int bc = n * 16 + cl;
                fb[n].q[0] = *(const uint4*)&Bs[buf][bc][k0l + 8 * hl];
                fb[n].q[1] = *(const uint4*)&Bs[buf][bc][k0l + 16 + 8 * hl];
            }
            // hard scheduling fence: keep all 18 ds_load_b128 ahead of the
            // WMMA chain so the backend uses partial s_wait_dscnt counts and
            // issues the 8 WMMAs back-to-back.
            __builtin_amdgcn_sched_barrier(0);
#pragma unroll
            for (int n = 0; n < 8; ++n) {
                acc[n] = __builtin_amdgcn_wmma_f32_16x16x32_bf16(
                    false, fa.bf, false, fb[n].bf, (short)0, acc[n], false, false);
            }
            __builtin_amdgcn_sched_barrier(0);
        }

        if (have_next) {
            commit(buf ^ 1, nkb);              // write the *other* buffer
            __syncthreads();                   // one barrier per K panel
            buf ^= 1;
        }
    }

    // ---- epilogue: bias + relu, scatter by original token index ----
    int tokv[8];
#pragma unroll
    for (int v = 0; v < 8; ++v) {
        const int rloc = m0 + v + 8 * hl;      // C/D layout: M = v + 8*(lane>=16)
        tokv[v] = stok[rowStart + (rloc < rows ? rloc : 0)];
    }
#pragma unroll
    for (int n = 0; n < 8; ++n) {
        const int col = n0blk + n * 16 + cl;
        const float bv = bias[(size_t)expert * HID + col];
#pragma unroll
        for (int v = 0; v < 8; ++v) {
            const int rloc = m0 + v + 8 * hl;
            if (rloc < rows) {
                float val = acc[n][v] + bv;
                val = val > 0.f ? val : 0.f;
                if (LAYER1) hout[(size_t)tokv[v] * HID + col] = f2bf(val);
                else        yout[(size_t)tokv[v] * HID + col] = val;
            }
        }
    }
}

// ---------------- launcher ----------------
extern "C" void kernel_launch(void* const* d_in, const int* in_sizes, int n_in,
                              void* d_out, int out_size, void* d_ws, size_t ws_size,
                              hipStream_t stream) {
    (void)in_sizes; (void)n_in; (void)out_size; (void)ws_size;
    const float* x    = (const float*)d_in[0];
    const int*   ops  = (const int*)d_in[1];
    const float* oemb = (const float*)d_in[2];
    const float* W1   = (const float*)d_in[3];
    const float* b1   = (const float*)d_in[4];
    const float* W2   = (const float*)d_in[5];
    const float* b2   = (const float*)d_in[6];
    float* out = (float*)d_out;

    char* ws = (char*)d_ws;
    int*  counts = (int*)(ws + 0);
    int*  cursor = (int*)(ws + 64);
    int*  ntiles = (int*)(ws + 128);
    int4* tdesc  = (int4*)(ws + 256);
    int*  stok   = (int*)(ws + 4096);
    unsigned short* embb = (unsigned short*)(ws + 69632);
    unsigned short* w1t  = (unsigned short*)(ws + 131072);                 // 24 MB
    unsigned short* w2t  = (unsigned short*)(ws + 131072 + 25165824);      // 16 MB
    unsigned short* hbuf = (unsigned short*)(ws + 131072 + 25165824 + 16777216); // 32 MB

    k_zero_counts<<<1, 32, 0, stream>>>(counts);
    k_hist<<<BATCH / 256, 256, 0, stream>>>(ops, counts);
    k_build<<<1, 1, 0, stream>>>(counts, cursor, tdesc, ntiles);
    k_scatter<<<BATCH / 256, 256, 0, stream>>>(ops, cursor, stok);
    k_cvt_emb<<<(NOPS * EMBD) / 256, 256, 0, stream>>>(oemb, embb);
    k_transpose_cvt<<<dim3(HID / 32, K1 / 32, NOPS), dim3(32, 8), 0, stream>>>(W1, w1t, K1, HID);
    k_transpose_cvt<<<dim3(HID / 32, HID / 32, NOPS), dim3(32, 8), 0, stream>>>(W2, w2t, HID, HID);

    k_expert_gemm<K1, true><<<dim3(MAX_TILES, HID / BN), 256, 0, stream>>>(
        x, embb, (const unsigned short*)nullptr, w1t, b1, stok, tdesc, ntiles,
        hbuf, (float*)nullptr);
    k_expert_gemm<HID, false><<<dim3(MAX_TILES, HID / BN), 256, 0, stream>>>(
        (const float*)nullptr, (const unsigned short*)nullptr, hbuf, w2t, b2,
        stok, tdesc, ntiles, (unsigned short*)nullptr, out);
}
